// CrossAttention_59330678227078
// MI455X (gfx1250) — compile-verified, matrix operand order checked by reading, not run on hardware
//
#include <hip/hip_runtime.h>

typedef __attribute__((ext_vector_type(16))) _Float16 v16h;
typedef __attribute__((ext_vector_type(4)))  _Float16 v4h;
typedef __attribute__((ext_vector_type(2)))  _Float16 v2h;
typedef __attribute__((ext_vector_type(8)))  float    v8f;

#define BB 32
#define NN 4096
#define CC 768
#define HH 12
#define HD 64
#define HPAD 16
#define SCALE 0.125f

#define CHUNK   64                  // tokens per block
#define NCHUNK  (NN / CHUNK)        // 64
#define NKSTEP  (CC / 32)           // 24 K-steps for the logits GEMM
#define XSP     776                 // padded LDS row stride (halfs)
#define LSP     68                  // padded logits row stride (floats)
#define PSP     72                  // padded p row stride (halfs)

// WMMA 16-bit A/B fragment K-index map (16x32 layout, wave32)
__device__ __forceinline__ int kmap(int kbase, int i) {
    return (i < 8) ? (kbase + i) : (16 + kbase + (i - 8));
}

// ---------------------------------------------------------------- K1: q = x[:,0,:] @ Wq
__global__ void k_qproj(const float* __restrict__ x, const float* __restrict__ Wq,
                        float* __restrict__ q) {
    int b = blockIdx.x;          // 32
    int j = threadIdx.x;         // 768
    const float* xr = x + (size_t)b * NN * CC;   // token 0
    float acc = 0.f;
    for (int c = 0; c < CC; ++c)
        acc = fmaf(xr[c], Wq[(size_t)c * CC + j], acc);
    q[(size_t)b * CC + j] = acc;
}

// ---------------------------------------------------------------- K2: w-tilde pre-packed in WMMA B-fragment order
// wtp[((b*NKSTEP + s)*32 + lane)*16 + i] = SCALE * sum_d Wk[c, h*64+d] * q[b, h*64+d]
//   with h = lane&15, kbase = (lane>>4)*8, c = s*32 + kmap(kbase, i)
__global__ void k_wtilde(const float* __restrict__ Wk, const float* __restrict__ q,
                         _Float16* __restrict__ wtp) {
    int idx  = blockIdx.x * blockDim.x + threadIdx.x;   // 32*24*32*16 = 393216
    int i    = idx & 15;
    int lane = (idx >> 4) & 31;
    int rem  = idx >> 9;
    int s    = rem % NKSTEP;
    int b    = rem / NKSTEP;
    int h     = lane & 15;
    int kbase = (lane >> 4) * 8;
    int c     = s * 32 + kmap(kbase, i);
    float acc = 0.f;
    if (h < HH) {
        const float* wkc = Wk + (size_t)c * CC + h * HD;
        const float* qb  = q  + (size_t)b * CC + h * HD;
        for (int d = 0; d < HD; ++d)
            acc = fmaf(wkc[d], qb[d], acc);
        acc *= SCALE;
    }
    wtp[idx] = (_Float16)acc;
}

// ---------------------------------------------------------------- K3: fused logits + chunk softmax + partial xbar (single read of x)
// blockDim = 128 (4 waves); ~106 KB LDS -> 3 blocks/WGP so streaming overlaps compute
__global__ void __launch_bounds__(128)
k_fused(const float* __restrict__ x, const _Float16* __restrict__ wtp,
        _Float16* __restrict__ part_xbar, float* __restrict__ part_m,
        float* __restrict__ part_l) {
    __shared__ _Float16 xs[CHUNK * XSP];      // x chunk, f16, padded rows (99328 B)
    __shared__ float    ls[HPAD * LSP];       // logits (4352 B)
    __shared__ _Float16 ps[HPAD * PSP];       // exp(l - m_loc) (2304 B)
    __shared__ float    mloc[HPAD];

    int b     = blockIdx.y;
    int chunk = blockIdx.x;
    int t0    = chunk * CHUNK;
    int tid   = threadIdx.x;
    int lane  = tid & 31;
    int wave  = tid >> 5;                     // 4 waves
    int m16   = lane & 15;
    int hi    = lane >> 4;
    int kbase = hi * 8;

    // ---- stage 1: global fp32 -> LDS f16 (coalesced float4 stream) ----
    const float* xbp = x + ((size_t)b * NN + t0) * CC;
    for (int i = tid; i < CHUNK * CC / 4; i += 128) {   // 96 iters
        float4 v = ((const float4*)xbp)[i];
        int t = i / (CC / 4);
        int c = (i % (CC / 4)) * 4;
        v4h hv; hv[0] = (_Float16)v.x; hv[1] = (_Float16)v.y;
                hv[2] = (_Float16)v.z; hv[3] = (_Float16)v.w;
        *(v4h*)&xs[t * XSP + c] = hv;
    }
    __syncthreads();

    // ---- stage 2: logits[h, t] via WMMA; wave w -> token tile w (4 tiles) ----
    {
        v8f acc = {};
        const _Float16* wbse = wtp + ((size_t)b * NKSTEP * 32 + lane) * 16;
        int arow = (wave * 16 + m16) * XSP;
        for (int s = 0; s < NKSTEP; ++s) {
            int kk = s * 32;
            v16h a;
#pragma unroll
            for (int p = 0; p < 8; ++p) {
                int k = (p < 4) ? (kbase + 2 * p) : (16 + kbase + 2 * (p - 4));
                v2h pr = *(const v2h*)&xs[arow + kk + k];
                a[2 * p] = pr[0]; a[2 * p + 1] = pr[1];
            }
            v16h bf = *(const v16h*)(wbse + (size_t)s * 32 * 16);  // 2x global_load_b128
            acc = __builtin_amdgcn_wmma_f32_16x16x32_f16(false, a, false, bf,
                                                         (short)0, acc, false, false);
        }
        // C/D: col = m16 (head), rows = 8*hi + r (token within tile)
#pragma unroll
        for (int r = 0; r < 8; ++r)
            ls[m16 * LSP + wave * 16 + 8 * hi + r] = acc[r];
    }
    __syncthreads();

    // ---- stage 3: per-head chunk stats (m_loc, l_loc) ----
    if (tid < HPAD) {
        int h = tid;
        float mx = 0.f, s = 0.f;
        if (h < HH) {
            mx = -1e30f;
            for (int t = 0; t < CHUNK; ++t) mx = fmaxf(mx, ls[h * LSP + t]);
            for (int t = 0; t < CHUNK; ++t) s += __expf(ls[h * LSP + t] - mx);
        }
        mloc[h] = mx;
        size_t pidx = ((size_t)(b * NCHUNK) + chunk) * HPAD + h;
        part_m[pidx] = mx; part_l[pidx] = s;
    }
    __syncthreads();

    for (int i = tid; i < HPAD * CHUNK; i += 128) {
        int h = i / CHUNK;
        int t = i % CHUNK;
        float p = (h < HH) ? __expf(ls[h * LSP + t] - mloc[h]) : 0.f;
        ps[h * PSP + t] = (_Float16)p;
    }
    __syncthreads();

    // ---- stage 4: partial xbar[h, c] = p @ x_chunk via WMMA (12 c-tiles/wave) ----
    _Float16* pout = part_xbar + ((size_t)(b * NCHUNK) + chunk) * HPAD * CC;
    for (int ct = wave; ct < CC / 16; ct += 4) {
        int c0 = ct * 16;
        v8f acc = {};
#pragma unroll
        for (int kt = 0; kt < CHUNK / 32; ++kt) {
            int kk = kt * 32;
            v16h a, bf;
#pragma unroll
            for (int p = 0; p < 8; ++p) {
                int k = (p < 4) ? (kbase + 2 * p) : (16 + kbase + 2 * (p - 4));
                v2h pr = *(const v2h*)&ps[m16 * PSP + kk + k];
                a[2 * p] = pr[0]; a[2 * p + 1] = pr[1];
            }
#pragma unroll
            for (int i = 0; i < 16; ++i)
                bf[i] = xs[(kk + kmap(kbase, i)) * XSP + c0 + m16];
            acc = __builtin_amdgcn_wmma_f32_16x16x32_f16(false, a, false, bf,
                                                         (short)0, acc, false, false);
        }
        // C/D: col = m16 (channel), rows = 8*hi + r (head)
#pragma unroll
        for (int r = 0; r < 8; ++r)
            pout[(size_t)(8 * hi + r) * CC + c0 + m16] = (_Float16)acc[r];
    }
}

// ---------------------------------------------------------------- K4: combine chunk partials -> xbar[b,h,c]
__global__ void k_combine(const _Float16* __restrict__ part_xbar,
                          const float* __restrict__ part_m,
                          const float* __restrict__ part_l,
                          float* __restrict__ xbar) {
    int idx = blockIdx.x * blockDim.x + threadIdx.x;   // 32*12*768 = 294912
    int c = idx % CC;
    int h = (idx / CC) % HH;
    int b = idx / (CC * HH);

    float M = -1e30f;
    for (int j = 0; j < NCHUNK; ++j)
        M = fmaxf(M, part_m[(size_t)(b * NCHUNK + j) * HPAD + h]);
    float L = 0.f, acc = 0.f;
    for (int j = 0; j < NCHUNK; ++j) {
        size_t pj = (size_t)(b * NCHUNK + j) * HPAD + h;
        float w = __expf(part_m[pj] - M);
        L   += w * part_l[pj];
        acc += w * (float)part_xbar[pj * CC + c];
    }
    xbar[((size_t)(b * HPAD) + h) * CC + c] = acc / L;
}

// ---------------------------------------------------------------- K5: o[b,h*64+d] = sum_c xbar[b,h,c] * Wv[c,h*64+d]
__global__ void k_oproj(const float* __restrict__ xbar, const float* __restrict__ Wv,
                        float* __restrict__ o) {
    int idx = blockIdx.x * blockDim.x + threadIdx.x;   // 24576
    int d = idx & (HD - 1);
    int h = (idx >> 6) % HH;
    int b = idx / (HD * HH);
    const float* xr = xbar + ((size_t)(b * HPAD) + h) * CC;
    const float* wv = Wv + h * HD + d;
    float acc = 0.f;
    for (int c = 0; c < CC; ++c)
        acc = fmaf(xr[c], wv[(size_t)c * CC], acc);
    o[(size_t)b * CC + h * HD + d] = acc;
}

// ---------------------------------------------------------------- K6: out[b,j] = bp[j] + sum_c o[b,c]*Wp[c,j]
__global__ void k_outproj(const float* __restrict__ o, const float* __restrict__ Wp,
                          const float* __restrict__ bp, float* __restrict__ out) {
    int b = blockIdx.x;      // 32
    int j = threadIdx.x;     // 768
    float acc = bp[j];
    const float* ob = o + (size_t)b * CC;
    for (int c = 0; c < CC; ++c)
        acc = fmaf(ob[c], Wp[(size_t)c * CC + j], acc);
    out[(size_t)b * CC + j] = acc;
}

// ---------------------------------------------------------------- launch
extern "C" void kernel_launch(void* const* d_in, const int* in_sizes, int n_in,
                              void* d_out, int out_size, void* d_ws, size_t ws_size,
                              hipStream_t stream) {
    const float* x  = (const float*)d_in[0];
    const float* Wq = (const float*)d_in[1];
    const float* Wk = (const float*)d_in[2];
    const float* Wv = (const float*)d_in[3];
    const float* Wp = (const float*)d_in[4];
    const float* bp = (const float*)d_in[5];
    float* out = (float*)d_out;

    char* ws = (char*)d_ws;
    float*    q         = (float*)   (ws + 0);          //    98304 B
    float*    o         = (float*)   (ws + 98304);      //    98304 B
    float*    xbar      = (float*)   (ws + 196608);     //  1572864 B
    _Float16* wtp       = (_Float16*)(ws + 1769472);    //   786432 B
    float*    part_m    = (float*)   (ws + 2555904);    //   131072 B
    float*    part_l    = (float*)   (ws + 2686976);    //   131072 B
    _Float16* part_xbar = (_Float16*)(ws + 2818048);    // 50331648 B -> total ~53.1 MB

    k_qproj  <<<dim3(BB),             dim3(CC),  0, stream>>>(x, Wq, q);
    k_wtilde <<<dim3(1536),           dim3(256), 0, stream>>>(Wk, q, wtp);
    k_fused  <<<dim3(NCHUNK, BB),     dim3(128), 0, stream>>>(x, wtp, part_xbar, part_m, part_l);
    k_combine<<<dim3(BB*HH*CC/256),   dim3(256), 0, stream>>>(part_xbar, part_m, part_l, xbar);
    k_oproj  <<<dim3(96),             dim3(256), 0, stream>>>(xbar, Wv, o);
    k_outproj<<<dim3(BB),             dim3(CC),  0, stream>>>(o, Wp, bp, out);
}
